// GRULanguageModel_30416958390393
// MI455X (gfx1250) — matrix operand order repaired and checked by reading, not compiled, not run on hardware
//
#include <hip/hip_runtime.h>
#include <hip/hip_bf16.h>

// Problem constants
#define BB 16
#define TT 256
#define EE 512
#define HH 1024
#define VV 32000
#define G3H 3072   // 3*H
#define G6H 6144   // 6*H

typedef __attribute__((ext_vector_type(8)))  _Float16 v8h;
typedef __attribute__((ext_vector_type(16))) _Float16 v16h;
typedef __attribute__((ext_vector_type(8)))  float    v8f;
typedef __attribute__((ext_vector_type(4)))  unsigned int u32x4;
typedef __attribute__((ext_vector_type(8)))  int      i32x8;
typedef __attribute__((ext_vector_type(4)))  int      i32x4;

static __device__ __forceinline__ v16h concat8(v8h lo, v8h hi) {
  return __builtin_shufflevector(lo, hi, 0,1,2,3,4,5,6,7,8,9,10,11,12,13,14,15);
}
static __device__ __forceinline__ v8f wmma_16x16x32(v16h a, v16h b, v8f c) {
  // (neg_a, A, neg_b, B, c_mod, C, reuse_a, reuse_b)
  return __builtin_amdgcn_wmma_f32_16x16x32_f16(false, a, false, b, (short)0, c, false, false);
}
static __device__ __forceinline__ float sigmoidf_(float v) {
  return 1.0f / (1.0f + __expf(-v));
}

// ---------------------------------------------------------------------------
// TDM 1-D panel load: copy n_qwords*8 bytes from global to LDS via the Tensor
// Data Mover. Descriptor per CDNA5 ISA §8.3/§8.4:
//  group0: [1:0]=count=1, [63:32]=lds_addr, [120:64]=global_addr, [127:126]=2
//  group1: [17:16]=data_size(3 => 8B), tensor_dim0[79:48], tile_dim0[127:112],
//          tensor_dim0_stride[207:160]; all other dims 0 (pure 1-D tile).
// Wave-level op (EXEC ignored); tracked by TENSORcnt.
// This toolchain exposes the 6-arg builtin: (g0, g1, g2, g3, g4, cpol).
// Groups 2..4 are for 3D+/gather descriptors; zero for a 1-D tile.
// ---------------------------------------------------------------------------
static __device__ __forceinline__ void tdm_load_1d(unsigned lds_off,
                                                   const void* gptr,
                                                   unsigned n_qwords) {
  unsigned long long ga = (unsigned long long)(uintptr_t)gptr;
  u32x4 g0;
  g0[0] = 1u;                                             // count=1, user mode
  g0[1] = lds_off;                                        // lds_addr (bytes)
  g0[2] = (unsigned)(ga & 0xFFFFFFFFu);                   // global_addr[31:0]
  g0[3] = (unsigned)((ga >> 32) & 0x01FFFFFFu) | (2u << 30); // addr[56:32]|type=2
  i32x8 g1;
  g1[0] = (int)(3u << 16);                                // data_size = 8 bytes
  g1[1] = (int)((n_qwords & 0xFFFFu) << 16);              // tensor_dim0 (lo bits)
  g1[2] = 0;                                              // tensor_dim0 hi / dim1
  g1[3] = (int)((n_qwords & 0xFFFFu) << 16);              // tile_dim0
  g1[4] = 0;                                              // tile_dim1/2 = 0
  g1[5] = (int)n_qwords;                                  // tensor_dim0_stride
  g1[6] = 0;
  g1[7] = 0;
  i32x4 z4 = {0, 0, 0, 0};
  i32x8 z8 = {0, 0, 0, 0, 0, 0, 0, 0};
  __builtin_amdgcn_tensor_load_to_lds(g0, g1, z4, z4, z8, 0);
}

// ---------------------------------------------------------------------------
// Init: zero block 0 of h0 sequence (h_{-1} = 0), fp32 + f16 copies.
// ---------------------------------------------------------------------------
__global__ void init_h0_kernel(float* __restrict__ h0f, _Float16* __restrict__ h0h) {
  int i = blockIdx.x * blockDim.x + threadIdx.x;
  if (i < BB * HH) { h0f[i] = 0.0f; h0h[i] = (_Float16)0.0f; }
}

// ---------------------------------------------------------------------------
// Embedding gather into time-major f16 matrix XE[t*B+b][E].
// ---------------------------------------------------------------------------
__global__ void gather_embed_kernel(const int* __restrict__ x,
                                    const float* __restrict__ emb,
                                    _Float16* __restrict__ XE) {
  int row = blockIdx.x;             // t*BB + b
  int t = row >> 4, b = row & 15;
  int tok = x[b * TT + t];
  const float* src = emb + (size_t)tok * EE;
  _Float16* dst = XE + (size_t)row * EE;
  for (int e = threadIdx.x; e < EE; e += blockDim.x) dst[e] = (_Float16)src[e];
}

// ---------------------------------------------------------------------------
// Pack fp32 weight W[K][Nsrc] (row-major) into f16 WMMA B-operand tiles.
// Tile (kt,nt): 512 halves; lane n holds, for elem i:
//   W[kt*32 + (i>>3)*16 + (n>>4)*8 + (i&7)][nt*16 + (n&15)]
// matching the ISA 16-bit B 32x16 layout. Dest tile index = kt*numNTdst+ntOff+nt.
// ---------------------------------------------------------------------------
__global__ void pack_w_kernel(const float* __restrict__ W, _Float16* __restrict__ P,
                              int K, int Nsrc, int numNTdst, int ntOff) {
  size_t idx = (size_t)blockIdx.x * blockDim.x + threadIdx.x;
  size_t total = (size_t)K * Nsrc;
  if (idx >= total) return;
  int elem = (int)(idx & 15);
  int lane = (int)((idx >> 4) & 31);
  size_t tile = idx >> 9;
  int numNTsrc = Nsrc >> 4;
  int kt = (int)(tile / numNTsrc);
  int nt = (int)(tile % numNTsrc);
  int kk = kt * 32 + ((elem >> 3) * 16) + ((lane >> 4) * 8) + (elem & 7);
  int nn = nt * 16 + (lane & 15);
  float v = W[(size_t)kk * Nsrc + nn];
  size_t dst = ((size_t)kt * numNTdst + ntOff + nt) * 512 + (size_t)lane * 16 + elem;
  P[dst] = (_Float16)v;
}

// ---------------------------------------------------------------------------
// bc1 = [bx1 ; bh1]  (6144 floats)
// ---------------------------------------------------------------------------
__global__ void concat_bias_kernel(const float* __restrict__ bx,
                                   const float* __restrict__ bh,
                                   float* __restrict__ bc) {
  int i = blockIdx.x * blockDim.x + threadIdx.x;
  if (i < G3H) { bc[i] = bx[i]; bc[G3H + i] = bh[i]; }
}

// ---------------------------------------------------------------------------
// f16 WMMA GEMM with TDM-staged B panels:
//   C[M][N] = A[M][K] @ Bpacked + bias (fp32 out, ldc = N).
// Block = 8 waves = one 128x128 C tile (wave w owns rows m0..m0+15, all 128 cols).
// Per K-step the 8 shared B tiles (8 KB, contiguous in packed layout) are
// DMA'd into LDS by the TDM (double buffered): wave0 issues tensor_load_to_lds
// for step k+32, everyone WMMAs out of the current buffer, wave0 waits
// TENSORcnt==0, workgroup barrier, swap.
// ---------------------------------------------------------------------------
__global__ __launch_bounds__(256)
void gemm_f16_tdm_kernel(const _Float16* __restrict__ A, const _Float16* __restrict__ Bp,
                         const float* __restrict__ bias, float* __restrict__ C,
                         int M, int N, int K) {
  __shared__ _Float16 bpanel[2][8 * 512];      // 2 x 8KB
  int lane = threadIdx.x & 31;
  int wid  = threadIdx.x >> 5;
  int mBlocks = M >> 7;
  int bm = blockIdx.x % mBlocks;
  int bn = blockIdx.x / mBlocks;
  int m0 = (bm << 7) + (wid << 4);
  int n0 = bn << 7;
  int numNT = N >> 4;
  int nt0 = n0 >> 4;

  int row = m0 + (lane & 15);
  int khalf = (lane >> 4) * 8;
  const _Float16* Arow = A + (size_t)row * K;

  v8f zero = {0.f,0.f,0.f,0.f,0.f,0.f,0.f,0.f};
  v8f acc[8];
#pragma unroll
  for (int j = 0; j < 8; ++j) acc[j] = zero;

  // Prologue: stage panel for k0 = 0 into buffer 0.
  if (wid == 0) {
    tdm_load_1d((unsigned)(uintptr_t)&bpanel[0][0],
                Bp + (size_t)nt0 * 512, 1024);
    __builtin_amdgcn_s_wait_tensorcnt(0);
  }
  __syncthreads();

  int cur = 0;
  for (int k0 = 0; k0 < K; k0 += 32) {
    bool more = (k0 + 32) < K;
    if (wid == 0 && more) {
      const _Float16* gp =
          Bp + ((size_t)((k0 + 32) >> 5) * numNT + nt0) * 512;
      tdm_load_1d((unsigned)(uintptr_t)&bpanel[cur ^ 1][0], gp, 1024);
    }
    v8h alo = *(const v8h*)(Arow + k0 + khalf);
    v8h ahi = *(const v8h*)(Arow + k0 + khalf + 16);
    v16h a = concat8(alo, ahi);
    const _Float16* bt = &bpanel[cur][(size_t)lane * 16];
#pragma unroll
    for (int j = 0; j < 8; ++j) {
      v16h b = *(const v16h*)(bt + (size_t)j * 512);
      acc[j] = wmma_16x16x32(a, b, acc[j]);
    }
    if (wid == 0 && more) __builtin_amdgcn_s_wait_tensorcnt(0);
    __syncthreads();
    cur ^= 1;
  }

  int rbase = m0 + ((lane >> 4) << 3);
  int cl = lane & 15;
#pragma unroll
  for (int j = 0; j < 8; ++j) {
    int col = n0 + j * 16 + cl;
    float bv = bias[col];
#pragma unroll
    for (int i = 0; i < 8; ++i) {
      C[(size_t)(rbase + i) * N + col] = acc[j][i] + bv;
    }
  }
}

// ---------------------------------------------------------------------------
// Layer-0 GRU step t (serial part). 64 blocks x 96 threads (3 waves).
// Wave w computes GH tile for gate w at columns w*H + [c0,c0+16), K=1024 WMMA,
// tiles meet in LDS, then fused gate math writes h0_seq block t+1 (f32 + f16).
// h0 buffers have T+1 blocks of [16][H]; block 0 is zeros.
// ---------------------------------------------------------------------------
__global__ __launch_bounds__(96)
void gru0_step_kernel(_Float16* __restrict__ h0h, float* __restrict__ h0f,
                      const _Float16* __restrict__ Wh0p,
                      const float* __restrict__ bh0,
                      const float* __restrict__ GX0, int t) {
  __shared__ float gh[3][16][16];
  int tid = threadIdx.x;
  int w = tid >> 5;                 // gate 0=r,1=z,2=n
  int lane = tid & 31;
  int c0 = blockIdx.x << 4;         // 0..1008
  int col0 = w * HH + c0;

  int row = t * BB + (lane & 15);
  int khalf = (lane >> 4) * 8;
  const _Float16* Arow = h0h + (size_t)row * HH;

  v8f acc = {0.f,0.f,0.f,0.f,0.f,0.f,0.f,0.f};
  const int numNT = G3H >> 4;       // 192
  for (int k0 = 0; k0 < HH; k0 += 32) {
    v8h alo = *(const v8h*)(Arow + k0 + khalf);
    v8h ahi = *(const v8h*)(Arow + k0 + khalf + 16);
    v16h a = concat8(alo, ahi);
    const _Float16* btile =
        Wh0p + ((size_t)(k0 >> 5) * numNT + (col0 >> 4)) * 512 + (size_t)lane * 16;
    v16h b = *(const v16h*)btile;
    acc = wmma_16x16x32(a, b, acc);
  }

  int rbase = (lane >> 4) << 3;
  int cl = lane & 15;
  float bv = bh0[col0 + cl];
#pragma unroll
  for (int i = 0; i < 8; ++i) gh[w][rbase + i][cl] = acc[i] + bv;
  __syncthreads();

  for (int idx = tid; idx < 256; idx += 96) {
    int r = idx >> 4, c = idx & 15;
    int col = c0 + c;
    size_t gxrow = (size_t)(t * BB + r) * G3H;
    float xr = GX0[gxrow + col];
    float xz = GX0[gxrow + HH + col];
    float xn = GX0[gxrow + 2 * HH + col];
    float rr = sigmoidf_(xr + gh[0][r][c]);
    float zz = sigmoidf_(xz + gh[1][r][c]);
    float nn = tanhf(xn + rr * gh[2][r][c]);
    float hp = h0f[(size_t)(t * BB + r) * HH + col];
    float hv = (1.0f - zz) * nn + zz * hp;
    size_t orow = (size_t)((t + 1) * BB + r) * HH + col;
    h0f[orow] = hv;
    h0h[orow] = (_Float16)hv;
  }
}

// ---------------------------------------------------------------------------
// Layer-1 gates (fully parallel): G1[row][6H] holds [gx|gh] for row = t*B+b.
// Writes out_seq f16 in [B,T,H] order for the FC GEMM; emits hidden at t=T-1.
// ---------------------------------------------------------------------------
__global__ void gru1_elem_kernel(const float* __restrict__ G1,
                                 const float* __restrict__ h0f,   // T+1 blocks
                                 _Float16* __restrict__ outseq,   // [B*T, H]
                                 float* __restrict__ hidden) {    // [2,16,H]
  size_t idx = (size_t)blockIdx.x * blockDim.x + threadIdx.x;
  if (idx >= (size_t)BB * TT * HH) return;
  int col = (int)(idx & (HH - 1));
  int row = (int)(idx >> 10);       // t*BB + b
  int t = row >> 4, b = row & 15;
  const float* g = G1 + (size_t)row * G6H;
  float xr = g[col], xz = g[HH + col], xn = g[2 * HH + col];
  float hr = g[3 * HH + col], hz = g[4 * HH + col], hn = g[5 * HH + col];
  float h0v = h0f[(size_t)(row + BB) * HH + col];  // h0_new at step t
  float rr = sigmoidf_(xr + hr);
  float zz = sigmoidf_(xz + hz);
  float nn = tanhf(xn + rr * hn);
  float h1 = (1.0f - zz) * nn + zz * h0v;
  outseq[((size_t)b * TT + t) * HH + col] = (_Float16)h1;
  if (t == TT - 1) {
    hidden[(size_t)b * HH + col] = h0v;                   // h0_final
    hidden[(size_t)BB * HH + (size_t)b * HH + col] = h1;  // h1_final
  }
}

// ---------------------------------------------------------------------------
extern "C" void kernel_launch(void* const* d_in, const int* in_sizes, int n_in,
                              void* d_out, int out_size, void* d_ws, size_t ws_size,
                              hipStream_t stream) {
  const int*   x   = (const int*)d_in[0];
  const float* emb = (const float*)d_in[1];
  const float* Wx0 = (const float*)d_in[2];
  const float* Wh0 = (const float*)d_in[3];
  const float* bx0 = (const float*)d_in[4];
  const float* bh0 = (const float*)d_in[5];
  const float* Wx1 = (const float*)d_in[6];
  const float* Wh1 = (const float*)d_in[7];
  const float* bx1 = (const float*)d_in[8];
  const float* bh1 = (const float*)d_in[9];
  const float* fcW = (const float*)d_in[10];
  const float* fcb = (const float*)d_in[11];

  float* logits = (float*)d_out;                       // [B,T,V]
  float* hidden = logits + (size_t)BB * TT * VV;       // [2,16,H]

  // ---- workspace layout (256B aligned) ----
  char* base = (char*)d_ws;
  size_t off = 0;
  auto alloc = [&](size_t bytes) -> char* {
    char* p = base + off;
    off = (off + bytes + 255) & ~(size_t)255;
    return p;
  };
  _Float16* XE    = (_Float16*)alloc((size_t)BB * TT * EE * 2);        // 4 MB
  _Float16* Wx0p  = (_Float16*)alloc((size_t)EE * G3H * 2);            // 3 MB
  _Float16* Wh0p  = (_Float16*)alloc((size_t)HH * G3H * 2);            // 6 MB
  _Float16* Wc1p  = (_Float16*)alloc((size_t)HH * G6H * 2);            // 12.6 MB
  _Float16* fcWp  = (_Float16*)alloc((size_t)HH * VV * 2);             // 65.5 MB
  float*    bc1   = (float*)alloc((size_t)G6H * 4);
  float*    GX0   = (float*)alloc((size_t)BB * TT * G3H * 4);          // 50 MB
  float*    h0f   = (float*)alloc((size_t)(TT + 1) * BB * HH * 4);     // 16.8 MB
  _Float16* h0h   = (_Float16*)alloc((size_t)(TT + 1) * BB * HH * 2);  // 8.4 MB
  float*    G1    = (float*)alloc((size_t)BB * TT * G6H * 4);          // 100 MB
  _Float16* outsq = (_Float16*)alloc((size_t)BB * TT * HH * 2);        // 8.4 MB

  // 1) init h_{-1} = 0
  init_h0_kernel<<<(BB * HH + 255) / 256, 256, 0, stream>>>(h0f, h0h);

  // 2) embedding gather (time-major rows)
  gather_embed_kernel<<<BB * TT, 256, 0, stream>>>(x, emb, XE);

  // 3) pack all weights into WMMA B layout
  {
    size_t n;
    n = (size_t)EE * G3H;
    pack_w_kernel<<<(unsigned)((n + 255) / 256), 256, 0, stream>>>(Wx0, Wx0p, EE, G3H, G3H >> 4, 0);
    n = (size_t)HH * G3H;
    pack_w_kernel<<<(unsigned)((n + 255) / 256), 256, 0, stream>>>(Wh0, Wh0p, HH, G3H, G3H >> 4, 0);
    pack_w_kernel<<<(unsigned)((n + 255) / 256), 256, 0, stream>>>(Wx1, Wc1p, HH, G3H, G6H >> 4, 0);
    pack_w_kernel<<<(unsigned)((n + 255) / 256), 256, 0, stream>>>(Wh1, Wc1p, HH, G3H, G6H >> 4, G3H >> 4);
    n = (size_t)HH * VV;
    pack_w_kernel<<<(unsigned)((n + 255) / 256), 256, 0, stream>>>(fcW, fcWp, HH, VV, VV >> 4, 0);
  }
  concat_bias_kernel<<<(G3H + 255) / 256, 256, 0, stream>>>(bx1, bh1, bc1);

  // 4) GX0 = XE @ Wx0 + bx0   [4096 x 3072]
  {
    int M = BB * TT, N = G3H, K = EE;
    int blocks = (M >> 7) * (N >> 7);
    gemm_f16_tdm_kernel<<<blocks, 256, 0, stream>>>(XE, Wx0p, bx0, GX0, M, N, K);
  }

  // 5) serial layer-0 scan: 256 dependent steps
  for (int t = 0; t < TT; ++t) {
    gru0_step_kernel<<<HH / 16, 96, 0, stream>>>(h0h, h0f, Wh0p, bh0, GX0, t);
  }

  // 6) G1 = h0_seq @ [Wx1|Wh1] + [bx1|bh1]   [4096 x 6144]
  {
    int M = BB * TT, N = G6H, K = HH;
    int blocks = (M >> 7) * (N >> 7);
    gemm_f16_tdm_kernel<<<blocks, 256, 0, stream>>>(h0h + (size_t)BB * HH, Wc1p, bc1, G1, M, N, K);
  }

  // 7) layer-1 gates + row remap to [B,T,H]; emit hidden
  {
    size_t n = (size_t)BB * TT * HH;
    gru1_elem_kernel<<<(unsigned)((n + 255) / 256), 256, 0, stream>>>(G1, h0f, outsq, hidden);
  }

  // 8) logits = out_seq @ fcW + fcb   [4096 x 32000] -> d_out
  {
    int M = BB * TT, N = VV, K = HH;
    int blocks = (M >> 7) * (N >> 7);
    gemm_f16_tdm_kernel<<<blocks, 256, 0, stream>>>(outsq, fcWp, fcb, logits, M, N, K);
  }
}
